// BSpline_7395933684402
// MI455X (gfx1250) — compile-verified, matrix operand order checked by reading, not run on hardware
//
#include <hip/hip_runtime.h>

typedef __attribute__((ext_vector_type(16))) _Float16 v16h;
typedef __attribute__((ext_vector_type(8)))  _Float16 v8h;
typedef __attribute__((ext_vector_type(8)))  float    v8f;

#define BATCH  8192
#define IN_F   512
#define OUT_F  512
#define NCOEF  8                 // GRID_SIZE + SPLINE_ORDER
#define KDIM   (IN_F * NCOEF)    // 4096
#define NKNOTS 12                // GRID_SIZE + 2*SPLINE_ORDER + 1

// ---------------------------------------------------------------------------
// CDNA5 async copy: global -> LDS, 16B per lane, tracked by ASYNCcnt.
// vdst = 32-bit LDS byte address (low 32 bits of generic shared pointer),
// vaddr = 64-bit global address, GV mode (saddr = off).
// ---------------------------------------------------------------------------
__device__ __forceinline__ void async_cp16(const _Float16* lds_dst,
                                           const _Float16* g_src) {
  uint32_t lds_addr = (uint32_t)(uintptr_t)lds_dst;
  asm volatile("global_load_async_to_lds_b128 %0, %1, off"
               :: "v"(lds_addr), "v"(g_src)
               : "memory");
}

template <short N>
__device__ __forceinline__ void wait_async_le() {
#if __has_builtin(__builtin_amdgcn_s_wait_asynccnt)
  __builtin_amdgcn_s_wait_asynccnt(N);   // template arg: integer constant
#else
  asm volatile("s_wait_asynccnt %0" :: "i"(N) : "memory");
#endif
}

// ---------------------------------------------------------------------------
// Kernel 1: bases[b, i*8+k] = cubic B-spline basis of tanh(x[b,i]), f16,
// row-major (BATCH x KDIM). One thread per (b,i); one 16B store per thread.
// ---------------------------------------------------------------------------
__global__ __launch_bounds__(256) void kan_bases(const float* __restrict__ x,
                                                 const float* __restrict__ grid,
                                                 _Float16* __restrict__ A) {
  const int idx = blockIdx.x * 256 + threadIdx.x;   // b*IN_F + i
  const float xn = tanhf(x[idx]);

  float t[NKNOTS];
#pragma unroll
  for (int j = 0; j < NKNOTS; ++j) t[j] = grid[j];  // uniform -> scalar loads

  // Cox-de Boor, degree 0..3
  float Bv[NKNOTS - 1];
#pragma unroll
  for (int j = 0; j < NKNOTS - 1; ++j)
    Bv[j] = (xn >= t[j] && xn < t[j + 1]) ? 1.0f : 0.0f;
#pragma unroll
  for (int k = 1; k <= 3; ++k) {
#pragma unroll
    for (int j = 0; j + k < NKNOTS - 1; ++j) {
      float dl = t[j + k] - t[j];
      dl = (dl == 0.0f) ? 1.0f : dl;
      float dr = t[j + k + 1] - t[j + 1];
      dr = (dr == 0.0f) ? 1.0f : dr;
      Bv[j] = (xn - t[j]) / dl * Bv[j] + (t[j + k + 1] - xn) / dr * Bv[j + 1];
    }
  }

  v8h o;
#pragma unroll
  for (int kk = 0; kk < NCOEF; ++kk) o[kk] = (_Float16)Bv[kk];
  *(v8h*)(A + (size_t)idx * NCOEF) = o;             // global_store_b128
}

// ---------------------------------------------------------------------------
// Kernel 2: repack coefficients[i, o, k] (f32) -> W[o, i*8+k] (f16): the B
// matrix in N-major order so the GEMM stages it exactly like A.
// ---------------------------------------------------------------------------
__global__ __launch_bounds__(256) void kan_pack_w(const float* __restrict__ coeff,
                                                  _Float16* __restrict__ W) {
  const int t = blockIdx.x * 256 + threadIdx.x;     // o*IN_F + i
  const int o = t >> 9;                             // /512
  const int i = t & (IN_F - 1);
  const float* src = coeff + ((size_t)i * OUT_F + o) * NCOEF;
  v8h w;
#pragma unroll
  for (int kk = 0; kk < NCOEF; ++kk) w[kk] = (_Float16)src[kk];
  *(v8h*)(W + (size_t)o * KDIM + i * NCOEF) = w;
}

// ---------------------------------------------------------------------------
// Kernel 3: C(8192x512, f32) = A(8192x4096, f16) x W^T (W stored N-major,
// 512x4096). 128x128 tile / block, K-step 32, 8 waves in 4(M) x 2(N); each
// wave: 2x4 grid of 16x16x32 WMMAs. Global->LDS via async copies with
// ASYNCcnt-counted double buffering.
// ---------------------------------------------------------------------------
#define MT   128
#define NT   128
#define KT   32
#define LDSS 40                 // 32 halfs + 8 pad: 80B row stride
#define BUFH (MT * LDSS)        // halfs per LDS buffer

__global__ __launch_bounds__(256) void kan_gemm(const _Float16* __restrict__ A,
                                                const _Float16* __restrict__ B,
                                                float* __restrict__ C) {
  __shared__ _Float16 As[2 * BUFH];
  __shared__ _Float16 Bs[2 * BUFH];

  const int m0   = blockIdx.y * MT;
  const int n0   = blockIdx.x * NT;
  const int tid  = threadIdx.x;
  const int lane = tid & 31;
  const int wave = tid >> 5;
  const int wm   = wave >> 1;           // 0..3 -> 32-row slab
  const int wn   = wave & 1;            // 0..1 -> 64-col slab
  const int r    = lane & 15;
  const int hi8  = (lane >> 4) * 8;     // K-half select per WMMA 16-bit layout

  // staging coords: 512 x 16B chunks per (A,B) tile, 2 chunks/thread/matrix
  const int row0 = tid >> 2;            // 0..63
  const int row1 = row0 + 64;           // 64..127
  const int off  = (tid & 3) * 8;

  const _Float16* gA0 = A + (size_t)(m0 + row0) * KDIM + off;
  const _Float16* gA1 = A + (size_t)(m0 + row1) * KDIM + off;
  const _Float16* gB0 = B + (size_t)(n0 + row0) * KDIM + off;
  const _Float16* gB1 = B + (size_t)(n0 + row1) * KDIM + off;

  const int lA0 = row0 * LDSS + off;
  const int lA1 = row1 * LDSS + off;

  v8f acc[2][4] = {};

  // prologue: tile 0 -> buffer 0
  async_cp16(As + lA0, gA0);
  async_cp16(As + lA1, gA1);
  async_cp16(Bs + lA0, gB0);
  async_cp16(Bs + lA1, gB1);

  for (int kb = 0, t = 0; kb < KDIM; kb += KT, ++t) {
    const int cur = t & 1;
    const int nxt = cur ^ 1;
    // issue next tile (wrap on last iter: harmless, keeps wait imm constant)
    const int kbn = (kb + KT < KDIM) ? (kb + KT) : 0;
    async_cp16(As + nxt * BUFH + lA0, gA0 + kbn);
    async_cp16(As + nxt * BUFH + lA1, gA1 + kbn);
    async_cp16(Bs + nxt * BUFH + lA0, gB0 + kbn);
    async_cp16(Bs + nxt * BUFH + lA1, gB1 + kbn);

    wait_async_le<4>();      // current tile's 4 copies complete (in-order)
    __syncthreads();         // ... across all 8 waves

    const _Float16* Ab = As + cur * BUFH;
    const _Float16* Bb = Bs + cur * BUFH;

    v16h af[2], bf[4];
#pragma unroll
    for (int sm = 0; sm < 2; ++sm) {
      const _Float16* p = Ab + (wm * 32 + sm * 16 + r) * LDSS;
      v8h lo = *(const v8h*)(p + hi8);        // K = hi8 .. hi8+7
      v8h hh = *(const v8h*)(p + 16 + hi8);   // K = 16+hi8 .. 16+hi8+7
      af[sm] = __builtin_shufflevector(lo, hh, 0, 1, 2, 3, 4, 5, 6, 7,
                                               8, 9, 10, 11, 12, 13, 14, 15);
    }
#pragma unroll
    for (int sn = 0; sn < 4; ++sn) {
      const _Float16* p = Bb + (wn * 64 + sn * 16 + r) * LDSS;
      v8h lo = *(const v8h*)(p + hi8);
      v8h hh = *(const v8h*)(p + 16 + hi8);
      bf[sn] = __builtin_shufflevector(lo, hh, 0, 1, 2, 3, 4, 5, 6, 7,
                                               8, 9, 10, 11, 12, 13, 14, 15);
    }
#pragma unroll
    for (int sm = 0; sm < 2; ++sm)
#pragma unroll
      for (int sn = 0; sn < 4; ++sn)
        acc[sm][sn] = __builtin_amdgcn_wmma_f32_16x16x32_f16(
            false, af[sm], false, bf[sn], (short)0, acc[sm][sn], false, false);

    __syncthreads();         // all waves done reading buf[cur] before reuse
  }

  // Epilogue: C layout — VGPR v holds row (v + 8*(lane>=16)), col = lane&15
#pragma unroll
  for (int sm = 0; sm < 2; ++sm) {
#pragma unroll
    for (int sn = 0; sn < 4; ++sn) {
      const int mbase = m0 + wm * 32 + sm * 16 + hi8;
      const int nidx  = n0 + wn * 64 + sn * 16 + r;
#pragma unroll
      for (int v = 0; v < 8; ++v)
        C[(size_t)(mbase + v) * OUT_F + nidx] = acc[sm][sn][v];
    }
  }
}

// ---------------------------------------------------------------------------
extern "C" void kernel_launch(void* const* d_in, const int* in_sizes, int n_in,
                              void* d_out, int out_size, void* d_ws, size_t ws_size,
                              hipStream_t stream) {
  (void)in_sizes; (void)n_in; (void)out_size; (void)ws_size;
  const float* x     = (const float*)d_in[0];   // 8192 x 512
  const float* coeff = (const float*)d_in[1];   // 512 x 512 x 8
  const float* grid  = (const float*)d_in[2];   // 12
  float* out = (float*)d_out;                   // 8192 x 512

  _Float16* Abases = (_Float16*)d_ws;                                   // 64 MB
  _Float16* W      = (_Float16*)((char*)d_ws +
                       (size_t)BATCH * KDIM * sizeof(_Float16));        // + 4 MB

  kan_bases <<<(BATCH * IN_F) / 256, 256, 0, stream>>>(x, grid, Abases);
  kan_pack_w<<<(OUT_F * IN_F) / 256, 256, 0, stream>>>(coeff, W);

  dim3 g(OUT_F / NT, BATCH / MT);   // 4 x 64 = 256 workgroups
  kan_gemm<<<g, 256, 0, stream>>>(Abases, W, out);
}
